// Conv3DSynthesisLayer_24773371363991
// MI455X (gfx1250) — compile-verified, matrix operand order checked by reading, not run on hardware
//
#include <hip/hip_runtime.h>

// ---------------------------------------------------------------------------
// Types for WMMA fragments (CDNA5 / gfx1250, wave32)
// ---------------------------------------------------------------------------
typedef __attribute__((ext_vector_type(16))) __bf16 v16bf;
typedef __attribute__((ext_vector_type(8)))  __bf16 v8bf;
typedef __attribute__((ext_vector_type(8)))  float  v8f;
typedef int v4i __attribute__((vector_size(16)));   // matches async-LDS builtin proto

#define LRELU_SLOPE 0.2f
#define ACT_GAIN    1.4142135623730951f
#define AFF_GAIN    0.04419417382415922f   // 1/sqrt(512)

// Problem constants
#define B_   8
#define CIN  128
#define COUT 128
#define R_   32
#define KOFF 27          // 3*3*3
#define KC   32          // Cin chunk (one WMMA K step)
#define NCHUNK (CIN / KC)

// ws layout: [0, 7077888) bf16 wmod in [b][kdh][chunk][kw][co][ci%32] order,
//            [7077888, +4096) f32 styles[b][ci]
#define WMOD_BYTES (size_t)(B_ * 9 * NCHUNK * 3 * COUT * KC * 2)
#define WGROUP_HALVES (3 * COUT * KC)     // 12288 halves = 24576 B per (b,kdh,chunk)

#define AS1 __attribute__((address_space(1)))
#define AS3 __attribute__((address_space(3)))

#if defined(__has_builtin)
#if __has_builtin(__builtin_amdgcn_global_load_async_to_lds_b128) && \
    __has_builtin(__builtin_amdgcn_s_wait_asynccnt)
#define USE_ASYNC_LDS 1
#endif
#endif

// ---------------------------------------------------------------------------
// Stage 1: styles[b][ci] = (w[b,:] . affine_weight[ci,:]) / sqrt(512) + bias
// ---------------------------------------------------------------------------
__global__ void styles_kernel(const float* __restrict__ w,
                              const float* __restrict__ aw,
                              const float* __restrict__ ab,
                              float* __restrict__ styles) {
    const int b  = blockIdx.x;      // 8
    const int ci = threadIdx.x;     // 128
    const float* wr  = w  + (size_t)b * 512;
    const float* awr = aw + (size_t)ci * 512;
    float s = 0.f;
    for (int k = 0; k < 512; ++k) s += wr[k] * awr[k];
    styles[b * CIN + ci] = s * AFF_GAIN + ab[ci];
}

// ---------------------------------------------------------------------------
// Stage 2: modulate + demodulate, emit bf16 weights in fragment-friendly order
//   wmod[(((((b*9+kdh)*4+chunk)*3+kw)*128+co)*32 + (ci%32)]
// ---------------------------------------------------------------------------
__global__ void modulate_kernel(const float* __restrict__ weight,
                                const float* __restrict__ styles,
                                __bf16* __restrict__ wmod) {
    const int co = blockIdx.x;      // 128
    const int b  = blockIdx.y;      // 8
    const int ci = threadIdx.x;     // 128
    const float st = styles[b * CIN + ci];
    const float* wp = weight + ((size_t)co * CIN + ci) * KOFF;

    float wv[KOFF];
    float ss = 0.f;
#pragma unroll
    for (int o = 0; o < KOFF; ++o) {
        float v = wp[o] * st;
        wv[o] = v;
        ss += v * v;
    }
    __shared__ float red[CIN];
    red[ci] = ss;
    __syncthreads();
#pragma unroll
    for (int s = 64; s > 0; s >>= 1) {
        if (ci < s) red[ci] += red[ci + s];
        __syncthreads();
    }
    const float dcoef = rsqrtf(red[0] + 1e-8f);

    const int chunk = ci >> 5;
    const int cil   = ci & 31;
#pragma unroll
    for (int o = 0; o < KOFF; ++o) {
        const int kdh = o / 3;          // compile-time under unroll
        const int kw  = o % 3;
        size_t idx = (((((size_t)b * 9 + kdh) * NCHUNK + chunk) * 3 + kw) * COUT + co) * KC + cil;
        wmod[idx] = (__bf16)(wv[o] * dcoef);
    }
}

// ---------------------------------------------------------------------------
// Fragment gather: 16 bf16 halves as two contiguous 16B LDS chunks.
// Per ISA §7.12.2 (16-bit A/B, wave32): lane-half hi selects K base 8*hi,
// elements 0..7 -> K in [8*hi, 8*hi+8), elements 8..15 -> K in [16+8*hi, ...).
// ---------------------------------------------------------------------------
__device__ __forceinline__ v16bf load_frag(const __bf16* row32, int kbase) {
    const v8bf lo = *reinterpret_cast<const v8bf*>(row32 + kbase);
    const v8bf hi = *reinterpret_cast<const v8bf*>(row32 + 16 + kbase);
    v16bf r;
#pragma unroll
    for (int i = 0; i < 8; ++i) { r[i] = lo[i]; r[i + 8] = hi[i]; }
    return r;
}

// ---------------------------------------------------------------------------
// Stage 3: implicit-GEMM conv. One block = (b, d, h): 128 Cout x 32 W outputs.
// 256 threads = 8 wave32; wave owns 16 Cout rows, two 16x16 N tiles.
// K loop: 4 Cin chunks x 9 (kd,kh) groups; each group = 3 kw offsets staged
// together -> 6 WMMAs per barrier pair.
// ---------------------------------------------------------------------------
__global__ __launch_bounds__(256)
void conv_kernel(const float* __restrict__ x,
                 const __bf16* __restrict__ wmod,
                 const float* __restrict__ bias,
                 float* __restrict__ out) {
    const int dh = blockIdx.x;          // 0..1023
    const int b  = blockIdx.y;          // 0..7
    const int d  = dh >> 5;
    const int h  = dh & 31;

    const int tid    = threadIdx.x;
    const int wave   = tid >> 5;        // 0..7 -> Cout tile
    const int lane   = tid & 31;
    const int m      = lane & 15;       // M-lane / N-lane index
    const int laneHi = lane >> 4;       // K-half select
    const int co0    = wave * 16;
    const int kbase  = laneHi * 8;

    // x tile: 9 halo rows x 34 padded W x 32 Cin (bf16), rows 16B aligned
    __shared__ __align__(16) __bf16 xl[9][34][KC];        // 19584 B
    __shared__ __align__(16) __bf16 wl[3][COUT][KC];      // 24576 B

    // zero the W-pad columns (wp = 0 and wp = 33) once
    for (int idx = tid; idx < 9 * 2 * KC; idx += 256) {
        int ci   = idx & 31;
        int r    = (idx >> 5) % 9;
        int side = idx / (KC * 9);
        xl[r][side ? 33 : 0][ci] = (__bf16)0.f;
    }

    v8f acc0 = {};   // N = 0..15
    v8f acc1 = {};   // N = 16..31

    const int wq = lane;                // x-staging W index (coalesced)

    for (int chunk = 0; chunk < NCHUNK; ++chunk) {
        const int ci0 = chunk * KC;
        // stage 9 halo rows of x for this Cin chunk (f32 -> bf16),
        // fully unrolled: (kd,kh) compile-time, bounds checks block-uniform.
#pragma unroll
        for (int r = 0; r < 9; ++r) {
            const int gd = d + (r / 3) - 1;
            const int gh = h + (r % 3) - 1;
            const bool in = (gd >= 0) & (gd < R_) & (gh >= 0) & (gh < R_);
            const float* xrow =
                x + ((((size_t)b * CIN + ci0) * R_ + (in ? gd : 0)) * R_ + (in ? gh : 0)) * R_ + wq;
#pragma unroll
            for (int c8 = 0; c8 < 4; ++c8) {
                const int ci = c8 * 8 + wave;
                float v = in ? xrow[(size_t)ci * (R_ * R_ * R_)] : 0.f;
                xl[r][wq + 1][ci] = (__bf16)v;
            }
        }

        for (int kdh = 0; kdh < 9; ++kdh) {
            // stage modulated weights for 3 kw offsets: [3][128 co][32 ci]
            {
                const __bf16* src =
                    wmod + (((size_t)b * 9 + kdh) * NCHUNK + chunk) * WGROUP_HALVES;
#if defined(USE_ASYNC_LDS)
                __bf16* dstb = &wl[0][0][0];
#pragma unroll
                for (int j = 0; j < 6; ++j) {
                    const int e = (tid + j * 256) * 8;   // halves (16B granules)
                    __builtin_amdgcn_global_load_async_to_lds_b128(
                        (AS1 v4i*)(src + e),
                        (AS3 v4i*)(dstb + e), 0, 0);
                }
                __builtin_amdgcn_s_wait_asynccnt(0);
#else
                const uint4* s4 = reinterpret_cast<const uint4*>(src);
                uint4* d4 = reinterpret_cast<uint4*>(&wl[0][0][0]);
#pragma unroll
                for (int j = 0; j < 6; ++j) d4[tid + j * 256] = s4[tid + j * 256];
#endif
            }
            __syncthreads();

            // A: 16 co x 32 ci per kw ; B: 32 ci x 16 w, two N tiles per kw
            v16bf a0 = load_frag(&wl[0][co0 + m][0], kbase);
            v16bf a1 = load_frag(&wl[1][co0 + m][0], kbase);
            v16bf a2 = load_frag(&wl[2][co0 + m][0], kbase);
            v16bf b00 = load_frag(&xl[kdh][m + 0][0], kbase);
            v16bf b01 = load_frag(&xl[kdh][m + 1][0], kbase);
            v16bf b02 = load_frag(&xl[kdh][m + 2][0], kbase);
            v16bf b10 = load_frag(&xl[kdh][m + 16][0], kbase);
            v16bf b11 = load_frag(&xl[kdh][m + 17][0], kbase);
            v16bf b12 = load_frag(&xl[kdh][m + 18][0], kbase);

            acc0 = __builtin_amdgcn_wmma_f32_16x16x32_bf16(
                false, a0, false, b00, (short)0, acc0, false, false);
            acc1 = __builtin_amdgcn_wmma_f32_16x16x32_bf16(
                false, a0, false, b10, (short)0, acc1, false, false);
            acc0 = __builtin_amdgcn_wmma_f32_16x16x32_bf16(
                false, a1, false, b01, (short)0, acc0, false, false);
            acc1 = __builtin_amdgcn_wmma_f32_16x16x32_bf16(
                false, a1, false, b11, (short)0, acc1, false, false);
            acc0 = __builtin_amdgcn_wmma_f32_16x16x32_bf16(
                false, a2, false, b02, (short)0, acc0, false, false);
            acc1 = __builtin_amdgcn_wmma_f32_16x16x32_bf16(
                false, a2, false, b12, (short)0, acc1, false, false);
            __syncthreads();
        }
    }

    // Epilogue: C/D layout — VGPR v, laneHi: M = v + 8*laneHi, N = lane&15
    const size_t obase = (((size_t)b * COUT) * R_ + d) * R_ * R_ + (size_t)h * R_;
#pragma unroll
    for (int v = 0; v < 8; ++v) {
        const int co = co0 + v + laneHi * 8;
        const float bv = bias[co];
        float t0 = acc0[v] + bv;
        t0 = (t0 >= 0.f ? t0 : t0 * LRELU_SLOPE) * ACT_GAIN;
        out[obase + (size_t)co * (R_ * R_ * R_) + m] = t0;
        float t1 = acc1[v] + bv;
        t1 = (t1 >= 0.f ? t1 : t1 * LRELU_SLOPE) * ACT_GAIN;
        out[obase + (size_t)co * (R_ * R_ * R_) + 16 + m] = t1;
    }
}

// ---------------------------------------------------------------------------
extern "C" void kernel_launch(void* const* d_in, const int* in_sizes, int n_in,
                              void* d_out, int out_size, void* d_ws, size_t ws_size,
                              hipStream_t stream) {
    const float* x      = (const float*)d_in[0];   // [8,128,32,32,32]
    const float* w      = (const float*)d_in[1];   // [8,512]
    const float* weight = (const float*)d_in[2];   // [128,128,3,3,3]
    const float* aw     = (const float*)d_in[3];   // [128,512]
    const float* ab     = (const float*)d_in[4];   // [128]
    const float* bias   = (const float*)d_in[5];   // [128]
    float* out = (float*)d_out;

    __bf16* wmod  = (__bf16*)d_ws;
    float* styles = (float*)((char*)d_ws + WMOD_BYTES);

    styles_kernel<<<dim3(B_), dim3(CIN), 0, stream>>>(w, aw, ab, styles);
    modulate_kernel<<<dim3(COUT, B_), dim3(CIN), 0, stream>>>(weight, styles, wmod);
    conv_kernel<<<dim3(R_ * R_, B_), dim3(256), 0, stream>>>(x, wmod, bias, out);
}
